// SAAM_83623013253529
// MI455X (gfx1250) — matrix operand order repaired and checked by reading, not compiled, not run on hardware
//
#include <hip/hip_runtime.h>

// ---------------------------------------------------------------------------
// MI455X (gfx1250) implementation of the SAAM block:
//   proj(q,k,v,c) -> per-(bu,h) 320x320 attention -> +residual -> relayout
//   -> conv(1,1,7) -> relu -> conv(7,1,1) -> relu
// All contractions via v_wmma_f32_16x16x32_f16 (f16 in, f32 acc).
// f16 intermediates staged into LDS with gfx1250 async paths:
//   - global_load_async_to_lds_b128 (ASYNCcnt) for attention Q/K/V tiles
//   - tensor_load_to_lds (TDM, TENSORcnt) for the conv2 input slab
// Pipeline is memory bound (~150MB traffic vs ~22 GFLOP -> DRAM-limited at
// 23.3 TB/s), so data movement goes through LDS without VGPR round-trips.
// ---------------------------------------------------------------------------

typedef __attribute__((ext_vector_type(16))) _Float16 v16h;
typedef __attribute__((ext_vector_type(8)))  float    v8f;
typedef __attribute__((ext_vector_type(4)))  unsigned int v4u;
typedef __attribute__((ext_vector_type(8)))  unsigned int v8u;

#define BU   8
#define CHN  64
#define C8   8
#define V5   5
#define H64  64
#define W64  64
#define VW   320     // V5*W64
#define HW   4096    // H64*W64
#define VHW  20480   // V5*H64*W64
#define CO   40      // C8*ANG_OUT

static __device__ __forceinline__ v8f wmma16(v16h a, v16h b, v8f c) {
  return __builtin_amdgcn_wmma_f32_16x16x32_f16(
      /*neg_a=*/false, a, /*neg_b=*/false, b,
      /*c_mod=*/(short)0, c, /*reuse_a=*/false, /*reuse_b=*/false);
}

static __device__ __forceinline__ v8f zero8() {
  v8f z;
#pragma unroll
  for (int i = 0; i < 8; ++i) z[i] = 0.0f;
  return z;
}

// Async 16B copy global -> LDS (GV mode), tracked with ASYNCcnt.
static __device__ __forceinline__ void async_cp16(unsigned lds_off,
                                                  const void* gptr) {
  asm volatile("global_load_async_to_lds_b128 %0, %1, off"
               :: "v"(lds_off), "v"((unsigned long long)gptr)
               : "memory");
}
static __device__ __forceinline__ void async_wait0() {
  asm volatile("s_wait_asynccnt 0x0" ::: "memory");
}

// ---------------------------------------------------------------------------
// Kernel 1: fused pointwise projections. 96x64 weight GEMM over all spatial
// positions. Block = 256 thr (8 waves), each wave owns 16 positions.
// Outputs qh/kh [bu,8,h,320], vh/ch [bu,40,h,320] as f16 in attention layout.
// ---------------------------------------------------------------------------
__global__ __launch_bounds__(256) void k_proj(
    const float* __restrict__ buf,
    const float* __restrict__ Wq, const float* __restrict__ Wk,
    const float* __restrict__ Wv, const float* __restrict__ Wc,
    _Float16* __restrict__ qh, _Float16* __restrict__ kh,
    _Float16* __restrict__ vh, _Float16* __restrict__ ch)
{
  __shared__ _Float16 Wl[96][64];
  const int tid = threadIdx.x;

  for (int i = tid; i < 96 * 64; i += 256) {
    int o = i >> 6, c = i & 63;
    float w;
    if (o < 8)       w = Wq[o * 64 + c];
    else if (o < 16) w = Wk[(o - 8) * 64 + c];
    else if (o < 56) w = Wv[(o - 16) * 64 + c];
    else             w = Wc[(o - 56) * 64 + c];
    Wl[o][c] = (_Float16)w;
  }
  __syncthreads();

  const int wave  = tid >> 5;
  const int lane  = tid & 31;
  const int lhalf = lane >> 4;
  const int lidx  = lane & 15;
  const int gpos  = blockIdx.x * 128 + wave * 16;  // 16 positions per wave
  const int bu    = gpos / VHW;
  const int p     = gpos % VHW;                    // VHW % 16 == 0, no straddle

  const float* xbase = buf + (size_t)bu * CHN * VHW + p + lidx;
  __builtin_prefetch(xbase, 0, 0);  // gfx1250 global_prefetch_b8

  // B fragments: [32 ch x 16 pos], two k-steps cover 64 channels.
  v16h bfrag[2];
#pragma unroll
  for (int ks = 0; ks < 2; ++ks) {
    const int c0 = ks * 32 + lhalf * 16;
#pragma unroll
    for (int j = 0; j < 16; ++j)
      bfrag[ks][j] = (_Float16)xbase[(size_t)(c0 + j) * VHW];
  }

  v8f acc[6];
#pragma unroll
  for (int t = 0; t < 6; ++t) acc[t] = zero8();

#pragma unroll
  for (int ks = 0; ks < 2; ++ks) {
#pragma unroll
    for (int t = 0; t < 6; ++t) {
      v16h a;
      const int o  = t * 16 + lidx;            // A row M
      const int cb = ks * 32 + lhalf * 16;     // K base: contiguous 16 per half
#pragma unroll
      for (int j = 0; j < 8; ++j) a[j]     = Wl[o][cb + j];
#pragma unroll
      for (int j = 0; j < 8; ++j) a[8 + j] = Wl[o][cb + 8 + j];
      acc[t] = wmma16(a, bfrag[ks], acc[t]);
    }
  }

  // Epilogue: scatter D tiles. D: VGPR r -> M = r + lhalf*8, N = lidx.
  const int pc = p + lidx;
  const int v  = pc / HW;
  const int hh = (pc >> 6) & 63;
  const int w  = pc & 63;
  const int m  = v * 64 + w;
#pragma unroll
  for (int t = 0; t < 6; ++t) {
#pragma unroll
    for (int r = 0; r < 8; ++r) {
      const int o = t * 16 + r + lhalf * 8;
      const _Float16 val = (_Float16)acc[t][r];
      if (o < 8)
        qh[((size_t)(bu * 8 + o) * 64 + hh) * 320 + m] = val;
      else if (o < 16)
        kh[((size_t)(bu * 8 + (o - 8)) * 64 + hh) * 320 + m] = val;
      else if (o < 56)
        vh[((size_t)(bu * CO + (o - 16)) * 64 + hh) * 320 + m] = val;
      else
        ch[((size_t)(bu * CO + (o - 56)) * 64 + hh) * 320 + m] = val;
    }
  }
}

// ---------------------------------------------------------------------------
// Kernel 2: attention per (bu,h). Block = 128 thr (4 waves), each wave owns
// m-tiles of 16 rows. Q/K/V staged into LDS via async b128 copies (pure f16
// byte moves, no VGPR round-trip). S = Q^T K (c=8 padded to K=32), row
// softmax in LDS, O = P V^T, + Wc residual, write conv1 input layout.
// ---------------------------------------------------------------------------
__global__ __launch_bounds__(128) void k_attn(
    const _Float16* __restrict__ qh, const _Float16* __restrict__ kh,
    const _Float16* __restrict__ vh, const _Float16* __restrict__ ch,
    _Float16* __restrict__ x1)
{
  extern __shared__ char smem[];
  _Float16* Qs = (_Float16*)smem;         // [8][320]   @ 0
  _Float16* Ks = Qs + 8 * 320;            // [8][320]   @ 5120
  _Float16* Vs = Ks + 8 * 320;            // [40][320]  @ 10240
  float*    Ss = (float*)(Vs + 40 * 320); // [4 waves][16][320]

  const int tid = threadIdx.x;
  const int bu  = blockIdx.x >> 6;
  const int hh  = blockIdx.x & 63;

  // Async stage Q/K (320 chunks of 16B each) and V (1600 chunks).
  // Row c is 640B in both global and LDS => chunk i: row i/40, byte (i%40)*16.
  for (int i = tid; i < 320; i += 128) {
    const int c = i / 40, o = (i % 40) * 8;
    async_cp16((unsigned)(i * 16),
               qh + ((size_t)(bu * 8 + c) * 64 + hh) * 320 + o);
    async_cp16((unsigned)(5120 + i * 16),
               kh + ((size_t)(bu * 8 + c) * 64 + hh) * 320 + o);
  }
  for (int i = tid; i < 1600; i += 128) {
    const int c = i / 40, o = (i % 40) * 8;
    async_cp16((unsigned)(10240 + i * 16),
               vh + ((size_t)(bu * CO + c) * 64 + hh) * 320 + o);
  }
  async_wait0();
  __syncthreads();

  const int wave  = tid >> 5;
  const int lane  = tid & 31;
  const int lhalf = lane >> 4;
  const int lidx  = lane & 15;
  float* Sw = Ss + wave * 16 * 320;

  for (int mt = wave; mt < 20; mt += 4) {
    const int m0 = mt * 16;

    // A fragment: rows = m (16), K = channel (8 real, 24 zero-pad)
    v16h afr;
#pragma unroll
    for (int j = 0; j < 16; ++j) afr[j] = (_Float16)0;
#pragma unroll
    for (int j = 0; j < 8; ++j)
      afr[j] = (lhalf == 0) ? Qs[j * 320 + m0 + lidx] : (_Float16)0;

    // Pass 1: S = Q^T K, spill 16x320 f32 block to LDS
    for (int nt = 0; nt < 20; ++nt) {
      const int n0 = nt * 16;
      v16h bfr;
#pragma unroll
      for (int j = 0; j < 16; ++j) bfr[j] = (_Float16)0;
#pragma unroll
      for (int j = 0; j < 8; ++j)
        bfr[j] = (lhalf == 0) ? Ks[j * 320 + n0 + lidx] : (_Float16)0;
      v8f d = wmma16(afr, bfr, zero8());
#pragma unroll
      for (int r = 0; r < 8; ++r)
        Sw[(r + lhalf * 8) * 320 + n0 + lidx] = d[r];
    }

    // Row softmax: lane pair (lidx, lidx+16) splits 320 cols into 2x160
    {
      const int row = lidx;
      const int sb  = lhalf * 160;
      float mx = -3.0e30f;
      for (int j = 0; j < 160; ++j)
        mx = fmaxf(mx, Sw[row * 320 + sb + j]);
      mx = fmaxf(mx, __shfl_xor(mx, 16, 32));
      float sum = 0.0f;
      for (int j = 0; j < 160; ++j) {
        const float e = __expf(Sw[row * 320 + sb + j] - mx);
        Sw[row * 320 + sb + j] = e;
        sum += e;
      }
      sum += __shfl_xor(sum, 16, 32);
      const float inv = 1.0f / sum;
      for (int j = 0; j < 160; ++j)
        Sw[row * 320 + sb + j] *= inv;
    }

    // Pass 2: O[m, c] = sum_n P[m,n] V[c,n], n in steps of 32, c tiles of 16
    v8f accO[3];
#pragma unroll
    for (int t = 0; t < 3; ++t) accO[t] = zero8();

    for (int ns = 0; ns < 10; ++ns) {
      const int n0 = ns * 32;
      v16h pfr;  // A: rows m, K = n
#pragma unroll
      for (int j = 0; j < 8; ++j)
        pfr[j]     = (_Float16)Sw[lidx * 320 + n0 + lhalf * 16 + j];
#pragma unroll
      for (int j = 0; j < 8; ++j)
        pfr[8 + j] = (_Float16)Sw[lidx * 320 + n0 + lhalf * 16 + 8 + j];
#pragma unroll
      for (int ct = 0; ct < 3; ++ct) {
        const int c = ct * 16 + lidx;  // B col N
        v16h vfr;
#pragma unroll
        for (int k = 0; k < 16; ++k)
          vfr[k] = (c < CO) ? Vs[c * 320 + n0 + lhalf * 16 + k] : (_Float16)0;
        accO[ct] = wmma16(pfr, vfr, accO[ct]);
      }
    }

    // Epilogue: + residual, relayout to conv1 input
#pragma unroll
    for (int ct = 0; ct < 3; ++ct) {
      const int c = ct * 16 + lidx;  // out channel (c8*5 + ar)
      if (c < CO) {
        const int c8i = c / 5, ar = c % 5;
#pragma unroll
        for (int r = 0; r < 8; ++r) {
          const int m = m0 + r + lhalf * 8;
          const int v = m >> 6, w = m & 63;
          const float val =
              accO[ct][r] +
              (float)ch[((size_t)(bu * CO + c) * 64 + hh) * 320 + m];
          x1[(((size_t)(bu * CO + c8i * 5 + v) * 5 + ar) * 64 + hh) * 64 + w] =
              (_Float16)val;
        }
      }
    }
  }
}

// ---------------------------------------------------------------------------
// Kernel 3: conv(1,1,7) 40->64 + relu as im2col GEMM along W.
// Block = 128 thr (4 waves) per (bu, ar, h) row; reduction 280 -> 9x32 padded.
// ---------------------------------------------------------------------------
__global__ __launch_bounds__(128) void k_conv1(
    const float* __restrict__ W1, const _Float16* __restrict__ x1,
    _Float16* __restrict__ y1)
{
  __shared__ _Float16 W1s[64][288];  // zero-padded reduction
  __shared__ _Float16 Xr[42][72];    // halo row, rows 40..41 zero

  const int tid = threadIdx.x;
  const int b   = blockIdx.x;
  const int bu  = b / 320;
  const int ar  = (b / 64) % 5;
  const int hh  = b & 63;

  for (int i = tid; i < 64 * 288; i += 128) {
    const int o = i / 288, red = i % 288;
    W1s[o][red] = (_Float16)((red < 280) ? W1[o * 280 + red] : 0.0f);
  }
  for (int i = tid; i < 42 * 70; i += 128) {
    const int ic = i / 70, j = i % 70;
    const int w = j - 3;
    float val = 0.0f;
    if (ic < CO && w >= 0 && w < 64)
      val = (float)x1[(((size_t)(bu * CO + ic) * 5 + ar)) * HW + hh * 64 + w];
    Xr[ic][j] = (_Float16)val;
  }
  __syncthreads();

  const int wave  = tid >> 5;
  const int lane  = tid & 31;
  const int lhalf = lane >> 4;
  const int lidx  = lane & 15;
  const int w0    = wave * 16;

  v8f acc[4];
#pragma unroll
  for (int t = 0; t < 4; ++t) acc[t] = zero8();

  for (int ks = 0; ks < 9; ++ks) {
    v16h bfr;  // B: [32 red x 16 w]
    const int kb = ks * 32 + lhalf * 16;
#pragma unroll
    for (int k = 0; k < 16; ++k) {
      const int red = kb + k;
      const int ic = red / 7, t = red % 7;
      bfr[k] = Xr[ic][w0 + lidx + t];
    }
#pragma unroll
    for (int ot = 0; ot < 4; ++ot) {
      v16h afr;
      const int o  = ot * 16 + lidx;
      const int cb = ks * 32 + lhalf * 16;
#pragma unroll
      for (int j = 0; j < 8; ++j) afr[j]     = W1s[o][cb + j];
#pragma unroll
      for (int j = 0; j < 8; ++j) afr[8 + j] = W1s[o][cb + 8 + j];
      acc[ot] = wmma16(afr, bfr, acc[ot]);
    }
  }

  const int w = w0 + lidx;
#pragma unroll
  for (int ot = 0; ot < 4; ++ot) {
#pragma unroll
    for (int r = 0; r < 8; ++r) {
      const int o = ot * 16 + r + lhalf * 8;
      const float val = fmaxf(acc[ot][r], 0.0f);
      y1[(((size_t)(bu * 64 + o) * 5 + ar)) * HW + hh * 64 + w] = (_Float16)val;
    }
  }
}

// ---------------------------------------------------------------------------
// Kernel 4: conv(7,1,1) 64->64 + relu along D (=ar, size 5, pad 3).
// Block = 128 thr per (bu,h); reduction 64*7 = 448 = 14x32 exactly.
// Input slab Y1[bu, :, :, h, :] (320 lines x 128B, stride 8KB) fetched by the
// Tensor Data Mover into LDS (one 2D D#), overlapped with W2 f32->f16 convert.
// ---------------------------------------------------------------------------
__global__ __launch_bounds__(128) void k_conv2(
    const float* __restrict__ W2, const _Float16* __restrict__ y1,
    float* __restrict__ out)
{
  extern __shared__ char smem4[];
  _Float16* W2s = (_Float16*)smem4;      // [64][448]      @ 0
  _Float16* Ys  = W2s + 64 * 448;        // [64][5][64]    @ 57344

  const int tid = threadIdx.x;
  const int bu  = blockIdx.x >> 6;
  const int hh  = blockIdx.x & 63;
  const int wave  = tid >> 5;
  const int lane  = tid & 31;
  const int lhalf = lane >> 4;
  const int lidx  = lane & 15;

  // --- TDM: one 2D descriptor moves the whole 40KB Y1 slab into LDS. ---
  // Tile: dim0 = 64 elems (f16, 128B line), dim1 = 320 lines (ic*5+d),
  // line stride = 4096 elems (8KB). Issued by wave 0 only.
  if (wave == 0) {
    const unsigned long long gaddr =
        (unsigned long long)(const void*)(y1 + (size_t)bu * 320 * HW + hh * 64);
    v4u g0;
    g0[0] = 1u;                                   // count=1 (valid user D#)
    g0[1] = 57344u;                               // lds_addr = offsetof(Ys)
    g0[2] = (unsigned)(gaddr & 0xFFFFFFFFu);      // global_addr[31:0]
    g0[3] = (unsigned)((gaddr >> 32) & 0x1FFFFFFu) | (2u << 30);  // [56:32]|type=2
    v8u g1;
    g1[0] = (1u << 16);          // data_size=1 (2 bytes), no multicast/pad
    g1[1] = (64u << 16);         // tensor_dim0[15:0] = 64
    g1[2] = (320u << 16);        // tensor_dim0[31:16]=0 | tensor_dim1[15:0]=320
    g1[3] = (64u << 16);         // tensor_dim1[31:16]=0 | tile_dim0=64
    g1[4] = 320u;                // tile_dim1=320, tile_dim2=0
    g1[5] = 4096u;               // tensor_dim0_stride[31:0] = 4096 elems
    g1[6] = 0u;                  // stride0[47:32]=0, stride1[15:0]=0
    g1[7] = 0u;                  // stride1 rest = 0
    asm volatile("tensor_load_to_lds %0, %1" :: "s"(g0), "s"(g1) : "memory");
  }

  // Overlap: VALU converts W2 f32 -> f16 while the TDM streams Ys.
  for (int i = tid; i < 64 * 448; i += 128)
    W2s[i] = (_Float16)W2[i];

  if (wave == 0) __builtin_amdgcn_s_wait_tensorcnt(0);
  __syncthreads();

  for (int mt = wave; mt < 20; mt += 4) {
    const int m0 = mt * 16;
    v8f acc[4];
#pragma unroll
    for (int t = 0; t < 4; ++t) acc[t] = zero8();

    for (int ks = 0; ks < 14; ++ks) {
      v16h bfr;  // B: [32 red x 16 m]
      const int m = m0 + lidx;
      const int d = m >> 6, w = m & 63;
      const int kb = ks * 32 + lhalf * 16;
#pragma unroll
      for (int k = 0; k < 16; ++k) {
        const int red = kb + k;
        const int ic = red / 7, t = red % 7;
        const int dd = d + t - 3;
        bfr[k] = (dd >= 0 && dd < 5) ? Ys[ic * 320 + dd * 64 + w]
                                     : (_Float16)0;
      }
#pragma unroll
      for (int ot = 0; ot < 4; ++ot) {
        v16h afr;
        const int o  = ot * 16 + lidx;
        const int cb = ks * 32 + lhalf * 16;
#pragma unroll
        for (int j = 0; j < 8; ++j) afr[j]     = W2s[o * 448 + cb + j];
#pragma unroll
        for (int j = 0; j < 8; ++j) afr[8 + j] = W2s[o * 448 + cb + 8 + j];
        acc[ot] = wmma16(afr, bfr, acc[ot]);
      }
    }

    const int m = m0 + lidx;
    const int d = m >> 6, w = m & 63;
#pragma unroll
    for (int ot = 0; ot < 4; ++ot) {
#pragma unroll
      for (int r = 0; r < 8; ++r) {
        const int o = ot * 16 + r + lhalf * 8;
        out[(((size_t)(bu * 64 + o) * 5 + d)) * HW + hh * 64 + w] =
            fmaxf(acc[ot][r], 0.0f);
      }
    }
  }
}

// ---------------------------------------------------------------------------
extern "C" void kernel_launch(void* const* d_in, const int* in_sizes, int n_in,
                              void* d_out, int out_size, void* d_ws,
                              size_t ws_size, hipStream_t stream) {
  const float* buf = (const float*)d_in[0];
  const float* Wq  = (const float*)d_in[1];
  const float* Wk  = (const float*)d_in[2];
  const float* Wv  = (const float*)d_in[3];
  const float* Wc  = (const float*)d_in[4];
  const float* W1  = (const float*)d_in[5];
  const float* W2  = (const float*)d_in[6];
  float* out = (float*)d_out;

  char* ws = (char*)d_ws;
  _Float16* qh = (_Float16*)ws;  ws += (size_t)BU * 8 * 64 * 320 * 2;   // 2.62 MB
  _Float16* kh = (_Float16*)ws;  ws += (size_t)BU * 8 * 64 * 320 * 2;   // 2.62 MB
  _Float16* vh = (_Float16*)ws;  ws += (size_t)BU * CO * 64 * 320 * 2;  // 13.1 MB
  _Float16* ch = (_Float16*)ws;  ws += (size_t)BU * CO * 64 * 320 * 2;  // 13.1 MB
  _Float16* x1 = (_Float16*)ws;  ws += (size_t)BU * CO * 5 * HW * 2;    // 13.1 MB
  _Float16* y1 = (_Float16*)ws;  ws += (size_t)BU * 64 * 5 * HW * 2;    // 21.0 MB

  // 1) fused q/k/v/c projections: 163840 positions, 128 per block
  k_proj<<<1280, 256, 0, stream>>>(buf, Wq, Wk, Wv, Wc, qh, kh, vh, ch);

  // 2) attention per (bu,h): dynamic LDS = Q+K+V f16 + 4x16x320 f32 S blocks
  const size_t lds2 = (size_t)(8 * 320 + 8 * 320 + 40 * 320) * 2 +
                      (size_t)4 * 16 * 320 * 4;  // 117,760 B
  k_attn<<<BU * 64, 128, lds2, stream>>>(qh, kh, vh, ch, x1);

  // 3) conv (1,1,7) + relu: one block per (bu, ar, h)
  k_conv1<<<BU * 5 * 64, 128, 0, stream>>>(W1, x1, y1);

  // 4) conv (7,1,1) + relu: one block per (bu, h), dynamic LDS = 98,304 B
  const size_t lds4 = (size_t)(64 * 448 + 64 * 5 * 64) * 2;
  k_conv2<<<BU * 64, 128, lds4, stream>>>(W2, y1, out);
}
